// SVDDense_5970004541710
// MI455X (gfx1250) — matrix operand order repaired and checked by reading, not compiled, not run on hardware
//
#include <hip/hip_runtime.h>

typedef __bf16 bf16_t;
typedef __attribute__((ext_vector_type(16))) __bf16 v16bf;
typedef __attribute__((ext_vector_type(8)))  __bf16 v8bf;
typedef __attribute__((ext_vector_type(4)))  __bf16 v4bf;
typedef __attribute__((ext_vector_type(8)))  float  v8f;

#define B_DIM   8192
#define D_IN    4096
#define RANK    2048
#define UNITS   4096

#define BM 128
#define BN 128
#define BK 32
#define NTHREADS 256
#define LDSW 40   // padded LDS row stride in bf16 elems (80B = 20 banks -> conflict-free frag reads)

// ---- CDNA5 async global->LDS copy (CDNA5 ISA), tracked by ASYNCcnt ----
__device__ __forceinline__ void async_load_b128(unsigned lds_off, unsigned long long gaddr) {
  asm volatile("global_load_async_to_lds_b128 %0, %1, off"
               :: "v"(lds_off), "v"(gaddr) : "memory");
}
__device__ __forceinline__ void wait_asynccnt0() {
  asm volatile("s_wait_asynccnt 0x0" ::: "memory");
}

// A fragment (16x32 bf16, ISA 7.12.2): lane L = row (L&15); lanes<16 hold K {0..7,16..23},
// lanes>=16 hold K {8..15,24..31} -> two 16B chunks at hf*16B and 32B+hf*16B.
__device__ __forceinline__ v16bf fragA_from_lds(const bf16_t* row_base, int hf) {
  v8bf lo = *(const v8bf*)(row_base + hf * 8);
  v8bf hi = *(const v8bf*)(row_base + 16 + hf * 8);
  return __builtin_shufflevector(lo, hi, 0,1,2,3,4,5,6,7,8,9,10,11,12,13,14,15);
}

// B fragment (32x16 bf16): lane L = column (L&15); lanes<16 hold K {0..7,16..23},
// lanes>=16 hold K {8..15,24..31} -> two 16B chunks at hf*16B and 32B+hf*16B.
__device__ __forceinline__ v16bf fragB_from_lds(const bf16_t* row_base, int hf) {
  v8bf lo = *(const v8bf*)(row_base + hf * 8);
  v8bf hi = *(const v8bf*)(row_base + 16 + hf * 8);
  return __builtin_shufflevector(lo, hi, 0,1,2,3,4,5,6,7,8,9,10,11,12,13,14,15);
}

__device__ __forceinline__ v4bf cvt4(float4 v) {
  v4bf r;
  r[0] = (bf16_t)v.x; r[1] = (bf16_t)v.y; r[2] = (bf16_t)v.z; r[3] = (bf16_t)v.w;
  return r;
}

// ---------------- GEMM1: H[b, r] = bf16( (X @ U)[b, r] * S[r] ) ----------------
__global__ __launch_bounds__(NTHREADS)
void svd_gemm1(const float* __restrict__ X, const float* __restrict__ U,
               const float* __restrict__ S, bf16_t* __restrict__ H) {
  __shared__ __align__(16) bf16_t As[2][BM * LDSW];
  __shared__ __align__(16) bf16_t Bt[2][BN * LDSW];   // stored transposed: [n][k]

  const int tid   = threadIdx.x;
  const int lane  = tid & 31;
  const int wid   = tid >> 5;
  const int waveM = wid >> 1;        // 0..3 -> 32 rows each
  const int waveN = wid & 1;         // 0..1 -> 64 cols each
  const int l16   = lane & 15;
  const int hf    = (lane >> 4) & 1;

  const int m0 = blockIdx.y * BM;
  const int n0 = blockIdx.x * BN;

  v8f acc[2][4] = {};
  float4 ar[4];
  float4 br[4];

  const int KT = D_IN / BK;          // 128

  auto gload = [&](int kt) {
    const int k0 = kt * BK;
#pragma unroll
    for (int i = 0; i < 4; ++i) {    // X tile: 128 rows x 32 cols fp32
      int f = tid + i * NTHREADS;
      int row = f >> 3, kc = f & 7;
      ar[i] = *(const float4*)(X + (size_t)(m0 + row) * D_IN + k0 + kc * 4);
    }
#pragma unroll
    for (int i = 0; i < 4; ++i) {    // U tile: 32 k-rows x 128 n-cols fp32
      int f = tid + i * NTHREADS;
      int k = f >> 5, nc = f & 31;
      br[i] = *(const float4*)(U + (size_t)(k0 + k) * RANK + n0 + nc * 4);
    }
  };
  auto lstore = [&](int buf) {
#pragma unroll
    for (int i = 0; i < 4; ++i) {
      int f = tid + i * NTHREADS;
      int row = f >> 3, kc = f & 7;
      *(v4bf*)(&As[buf][row * LDSW + kc * 4]) = cvt4(ar[i]);
    }
#pragma unroll
    for (int i = 0; i < 4; ++i) {    // transpose U tile into Bt[n][k]
      int f = tid + i * NTHREADS;
      int k = f >> 5, nc = f & 31;
      v4bf b4 = cvt4(br[i]);
#pragma unroll
      for (int j = 0; j < 4; ++j)
        Bt[buf][(nc * 4 + j) * LDSW + k] = b4[j];
    }
  };

  gload(0);
  lstore(0);
  __syncthreads();

  for (int kt = 0; kt < KT; ++kt) {
    const int cur = kt & 1;
    if (kt + 1 < KT) gload(kt + 1);

    v16bf a[2], b[4];
#pragma unroll
    for (int fm = 0; fm < 2; ++fm)
      a[fm] = fragA_from_lds(&As[cur][(waveM * 32 + fm * 16 + l16) * LDSW], hf);
#pragma unroll
    for (int fn = 0; fn < 4; ++fn)
      b[fn] = fragB_from_lds(&Bt[cur][(waveN * 64 + fn * 16 + l16) * LDSW], hf);

#pragma unroll
    for (int fm = 0; fm < 2; ++fm)
#pragma unroll
      for (int fn = 0; fn < 4; ++fn)
        acc[fm][fn] = __builtin_amdgcn_wmma_f32_16x16x32_bf16(
            false, a[fm], false, b[fn], (short)0, acc[fm][fn], false, false);

    if (kt + 1 < KT) lstore((kt + 1) & 1);
    __syncthreads();
  }

  // Epilogue: scale columns by S, store bf16 intermediate.
#pragma unroll
  for (int fn = 0; fn < 4; ++fn) {
    const int n = n0 + waveN * 64 + fn * 16 + l16;
    const float s = S[n];
#pragma unroll
    for (int fm = 0; fm < 2; ++fm) {
      const int mb = m0 + waveM * 32 + fm * 16 + hf * 8;
#pragma unroll
      for (int r = 0; r < 8; ++r)
        H[(size_t)(mb + r) * RANK + n] = (bf16_t)(acc[fm][fn][r] * s);
    }
  }
}

// ---------------- GEMM2: out = relu( H @ V^T + bias ) ----------------
__global__ __launch_bounds__(NTHREADS)
void svd_gemm2(const bf16_t* __restrict__ H, const float* __restrict__ V,
               const float* __restrict__ bias, float* __restrict__ out) {
  __shared__ __align__(16) bf16_t As[2][BM * LDSW];
  __shared__ __align__(16) bf16_t Bt[2][BN * LDSW];   // Bt[n][k] = V[n][k] (V^T columns)

  const int tid   = threadIdx.x;
  const int lane  = tid & 31;
  const int wid   = tid >> 5;
  const int waveM = wid >> 1;
  const int waveN = wid & 1;
  const int l16   = lane & 15;
  const int hf    = (lane >> 4) & 1;

  const int m0 = blockIdx.y * BM;
  const int n0 = blockIdx.x * BN;

  v8f acc[2][4] = {};
  float4 br[4];

  const int KT = RANK / BK;          // 64

  // A tile (bf16 H) goes straight to LDS via async DMA: 128 rows x 64B, 2 x b128 per lane.
  // LDS aperture keeps the offset in addr[31:0] (ISA §10.2), so truncation yields the DS address.
  const unsigned asBase = (unsigned)(size_t)(&As[0][0]);
  const unsigned ABUF   = (unsigned)(BM * LDSW * sizeof(bf16_t));
  unsigned long long gA[2];
  unsigned lA[2];
#pragma unroll
  for (int i = 0; i < 2; ++i) {
    int f = tid + i * NTHREADS;
    int row = f >> 2, kc = f & 3;
    gA[i] = (unsigned long long)(const void*)(H + (size_t)(m0 + row) * RANK + kc * 8);
    lA[i] = asBase + (unsigned)((row * LDSW + kc * 8) * sizeof(bf16_t));
  }
  auto issueA = [&](int buf) {
#pragma unroll
    for (int i = 0; i < 2; ++i) async_load_b128(lA[i] + (unsigned)buf * ABUF, gA[i]);
#pragma unroll
    for (int i = 0; i < 2; ++i) gA[i] += BK * sizeof(bf16_t);
  };

  auto gloadB = [&](int kt) {
    const int k0 = kt * BK;
#pragma unroll
    for (int i = 0; i < 4; ++i) {    // V tile: rows are n, contiguous in k
      int f = tid + i * NTHREADS;
      int n = f >> 3, kc = f & 7;
      br[i] = *(const float4*)(V + (size_t)(n0 + n) * RANK + k0 + kc * 4);
    }
  };
  auto lstoreB = [&](int buf) {
#pragma unroll
    for (int i = 0; i < 4; ++i) {
      int f = tid + i * NTHREADS;
      int n = f >> 3, kc = f & 7;
      *(v4bf*)(&Bt[buf][n * LDSW + kc * 4]) = cvt4(br[i]);
    }
  };

  issueA(0);
  gloadB(0);
  lstoreB(0);
  wait_asynccnt0();
  __syncthreads();

  for (int kt = 0; kt < KT; ++kt) {
    const int cur = kt & 1;
    if (kt + 1 < KT) {
      issueA((kt + 1) & 1);          // DMA next A tile while we compute
      gloadB(kt + 1);
    }

    v16bf a[2], b[4];
#pragma unroll
    for (int fm = 0; fm < 2; ++fm)
      a[fm] = fragA_from_lds(&As[cur][(waveM * 32 + fm * 16 + l16) * LDSW], hf);
#pragma unroll
    for (int fn = 0; fn < 4; ++fn)
      b[fn] = fragB_from_lds(&Bt[cur][(waveN * 64 + fn * 16 + l16) * LDSW], hf);

#pragma unroll
    for (int fm = 0; fm < 2; ++fm)
#pragma unroll
      for (int fn = 0; fn < 4; ++fn)
        acc[fm][fn] = __builtin_amdgcn_wmma_f32_16x16x32_bf16(
            false, a[fm], false, b[fn], (short)0, acc[fm][fn], false, false);

    if (kt + 1 < KT) lstoreB((kt + 1) & 1);
    wait_asynccnt0();                // async A-tile writes visible before the publish barrier
    __syncthreads();
  }

  // Epilogue: bias + ReLU, fp32 out.
#pragma unroll
  for (int fn = 0; fn < 4; ++fn) {
    const int n = n0 + waveN * 64 + fn * 16 + l16;
    const float bv = bias[n];
#pragma unroll
    for (int fm = 0; fm < 2; ++fm) {
      const int mb = m0 + waveM * 32 + fm * 16 + hf * 8;
#pragma unroll
      for (int r = 0; r < 8; ++r)
        out[(size_t)(mb + r) * UNITS + n] = fmaxf(acc[fm][fn][r] + bv, 0.0f);
    }
  }
}

extern "C" void kernel_launch(void* const* d_in, const int* in_sizes, int n_in,
                              void* d_out, int out_size, void* d_ws, size_t ws_size,
                              hipStream_t stream) {
  (void)in_sizes; (void)n_in; (void)out_size; (void)ws_size;
  const float* X    = (const float*)d_in[0];
  const float* U    = (const float*)d_in[1];
  const float* S    = (const float*)d_in[2];
  const float* V    = (const float*)d_in[3];
  const float* bias = (const float*)d_in[4];
  float* out = (float*)d_out;
  bf16_t* H = (bf16_t*)d_ws;   // 8192*2048 bf16 = 32 MiB intermediate

  dim3 g1(RANK / BN,  B_DIM / BM);   // (16, 64)
  svd_gemm1<<<g1, NTHREADS, 0, stream>>>(X, U, S, H);
  dim3 g2(UNITS / BN, B_DIM / BM);   // (32, 64)
  svd_gemm2<<<g2, NTHREADS, 0, stream>>>(H, V, bias, out);
}
